// Kge_finetune_87986700026237
// MI455X (gfx1250) — matrix operand order repaired and checked by reading, not compiled, not run on hardware
//
#include <hip/hip_runtime.h>
#include <hip/hip_bf16.h>

typedef __attribute__((ext_vector_type(16))) __bf16 v16bf;
typedef __attribute__((ext_vector_type(8)))  __bf16 v8bf;
typedef __attribute__((ext_vector_type(8)))  float  v8f;

#define RANK 256
#define DDIM 512
#define MAX_OBS 32
#define THRESH 1.0e-4f
#define EPS 1.0e-3f

// ---------------------------------------------------------------------------
// K1: compact indices of nonzeros in head_ent_vec (ascending), single block.
// ---------------------------------------------------------------------------
__global__ void __launch_bounds__(1024) find_heads(const float* __restrict__ hv,
                                                   int* __restrict__ heads,
                                                   int n_ent, int max_heads) {
  __shared__ int wsum[32];
  __shared__ int base;
  int tid = threadIdx.x;
  int lane = tid & 31, wave = tid >> 5;
  if (tid == 0) base = 0;
  __syncthreads();
  for (int start = 0; start < n_ent; start += 1024) {
    int i = start + tid;
    int flag = (i < n_ent && hv[i] != 0.0f) ? 1 : 0;
    unsigned long long bal = __ballot(flag);
    unsigned int b32 = (unsigned int)bal;
    int wprefix = __popc(b32 & ((1u << lane) - 1u));
    if (lane == 0) wsum[wave] = __popc(b32);
    __syncthreads();
    int woff = 0;
    for (int w = 0; w < wave; ++w) woff += wsum[w];
    int pos = base + woff + wprefix;
    if (flag && pos < max_heads) heads[pos] = i;
    __syncthreads();
    if (tid == 0) {
      int t = 0;
      for (int w = 0; w < 32; ++w) t += wsum[w];
      base += t;
    }
    __syncthreads();
  }
}

// ---------------------------------------------------------------------------
// K2: Q[h, 0:256] = re_h*re_r - im_h*im_r ; Q[h, 256:512] = re_h*im_r + im_h*re_r
//     stored as bf16 for the WMMA A-matrix.
// ---------------------------------------------------------------------------
__global__ void __launch_bounds__(RANK) build_q(const float* __restrict__ ent,
                                                const float* __restrict__ rel,
                                                const int* __restrict__ heads,
                                                const int* __restrict__ rel_id,
                                                __bf16* __restrict__ Q) {
  int hh = blockIdx.x;
  int k = threadIdx.x;               // 0..255
  int he = heads[hh];
  const float* r = rel + (size_t)rel_id[0] * DDIM;
  float re_h = ent[(size_t)he * DDIM + k];
  float im_h = ent[(size_t)he * DDIM + RANK + k];
  float re_r = r[k], im_r = r[RANK + k];
  Q[(size_t)hh * DDIM + k]        = (__bf16)(re_h * re_r - im_h * im_r);
  Q[(size_t)hh * DDIM + RANK + k] = (__bf16)(re_h * im_r + im_h * re_r);
}

// ---------------------------------------------------------------------------
// K3: score = Q (nh x 512, bf16) @ ent^T (512 x n_ent) via v_wmma_f32_16x16x32_bf16.
// One block = 16 entities (N-tile) x all heads. 8 waves, 2 M-tiles per wave.
// The 16x512 f32 entity tile is brought into LDS with the CDNA5 async
// global->LDS path (ASYNCcnt), then converted to bf16 once while building the
// (register-hoisted) B fragments.
// ---------------------------------------------------------------------------
__global__ void __launch_bounds__(256) gemm_score(const float* __restrict__ ent,
                                                  const __bf16* __restrict__ Q,
                                                  float* __restrict__ score,
                                                  int n_ent, int n_mtiles) {
  __shared__ __attribute__((aligned(16))) float fsm[16 * DDIM];  // 32 KB f32 tile
  int tid = threadIdx.x;
  int e0 = blockIdx.x * 16;

  // Async-copy 16 entity rows (8192 floats = 2048 x b128) straight into LDS.
#pragma unroll
  for (int i = 0; i < 8; ++i) {
    int f = tid + 256 * i;           // b128 index (128 per row)
    int row = f >> 7;
    int c4 = f & 127;
    const float* src = ent + (size_t)(e0 + row) * DDIM + c4 * 4;
    // Flat LDS address carries the byte offset in addr[31:0].
    unsigned ldsoff = (unsigned)(unsigned long long)(&fsm[row * DDIM + c4 * 4]);
    asm volatile("global_load_async_to_lds_b128 %0, %1, off"
                 :: "v"(ldsoff), "v"(src) : "memory");
  }
  asm volatile("s_wait_asynccnt 0" ::: "memory");
  __syncthreads();

  int lane = tid & 31, wave = tid >> 5;
  int kg = lane >> 4;                // half-wave selector
  int lm = lane & 15;

  // Build the 16 B fragments once (f32 -> bf16); they persist in VGPRs across
  // the M-tile loop. B layout: column n = lm, K = k0 + kg*16 + {0..15}.
  v16bf bfrag[16];
#pragma unroll
  for (int kk = 0; kk < 16; ++kk) {
    const float* bp = &fsm[lm * DDIM + kk * 32 + kg * 16];
#pragma unroll
    for (int i = 0; i < 16; ++i) bfrag[kk][i] = (__bf16)bp[i];
  }

  for (int mt = wave; mt < n_mtiles; mt += 8) {
    v8f c = {};
    const __bf16* qrow = Q + (size_t)(mt * 16 + lm) * DDIM;
#pragma unroll
    for (int kk = 0; kk < 16; ++kk) {
      int k0 = kk * 32;
      // A fragment: row lm, K = k0 + kg*16 + {0..15}
      v8bf a0 = *(const v8bf*)(qrow + k0 + kg * 16);
      v8bf a1 = *(const v8bf*)(qrow + k0 + kg * 16 + 8);
      v16bf a;
#pragma unroll
      for (int i = 0; i < 8; ++i) { a[i] = a0[i]; a[i + 8] = a1[i]; }
      c = __builtin_amdgcn_wmma_f32_16x16x32_bf16(false, a, false, bfrag[kk],
                                                  (short)0, c, false, false);
    }
    // C/D layout: n = lane%16, m = r + 8*(lane/16)
    float* outp = score + (size_t)(mt * 16) * n_ent + e0 + lm;
#pragma unroll
    for (int r = 0; r < 8; ++r) outp[(size_t)(r + 8 * kg) * n_ent] = c[r];
  }
}

// ---------------------------------------------------------------------------
// K4: per-head row max and sum(exp(s - max)) (two passes, score stays in L2).
// ---------------------------------------------------------------------------
__global__ void __launch_bounds__(256) softmax_stats(const float* __restrict__ score,
                                                     float* __restrict__ mx,
                                                     float* __restrict__ sm,
                                                     int n_ent) {
  __shared__ float red[256];
  int h = blockIdx.x, tid = threadIdx.x;
  const float* row = score + (size_t)h * n_ent;
  float lm = -3.4e38f;
  for (int e = tid; e < n_ent; e += 256) lm = fmaxf(lm, row[e]);
  red[tid] = lm; __syncthreads();
  for (int s = 128; s > 0; s >>= 1) {
    if (tid < s) red[tid] = fmaxf(red[tid], red[tid + s]);
    __syncthreads();
  }
  float m = red[0];
  __syncthreads();
  float ls = 0.0f;
  for (int e = tid; e < n_ent; e += 256) ls += __expf(row[e] - m);
  red[tid] = ls; __syncthreads();
  for (int s = 128; s > 0; s >>= 1) {
    if (tid < s) red[tid] += red[tid + s];
    __syncthreads();
  }
  if (tid == 0) { mx[h] = m; sm[h] = red[0]; }
}

// ---------------------------------------------------------------------------
// K5: scaling[h] = obs_num>0 ? obs_num / max(denom, 1e-30) : 1   (1 wave/head)
// ---------------------------------------------------------------------------
__global__ void __launch_bounds__(32) compute_scaling(const float* __restrict__ score,
                                                      const int* __restrict__ obs_idx,
                                                      const unsigned char* __restrict__ obs_mask,
                                                      const float* __restrict__ mx,
                                                      const float* __restrict__ sm,
                                                      float* __restrict__ scaling,
                                                      int n_ent) {
  int h = blockIdx.x, j = threadIdx.x;  // 32 lanes == MAX_OBS
  int msk = obs_mask[h * MAX_OBS + j] ? 1 : 0;
  int idx = obs_idx[h * MAX_OBS + j];
  float p = 0.0f;
  if (msk) p = __expf(score[(size_t)h * n_ent + idx] - mx[h]) / sm[h];
#pragma unroll
  for (int off = 16; off > 0; off >>= 1) {
    p += __shfl_down(p, off, 32);
    msk += __shfl_down(msk, off, 32);
  }
  if (j == 0) scaling[h] = (msk > 0) ? ((float)msk / fmaxf(p, 1e-30f)) : 1.0f;
}

// ---------------------------------------------------------------------------
// K6: out = clip(threshold(prob * scaling), 0, hi) in-place over the scores.
// ---------------------------------------------------------------------------
__global__ void __launch_bounds__(256) finalize(float* __restrict__ out,
                                                const float* __restrict__ mx,
                                                const float* __restrict__ sm,
                                                const float* __restrict__ scaling,
                                                const int* __restrict__ train_mask,
                                                int n_ent) {
  int h = blockIdx.y;
  int e = blockIdx.x * 256 + threadIdx.x;
  if (e >= n_ent) return;
  size_t i = (size_t)h * n_ent + e;
  float p = __expf(out[i] - mx[h]) / sm[h];
  float v = p * scaling[h];
  v = (v > THRESH) ? v : 0.0f;
  float hi = (train_mask[0] != 0) ? (1.0f - EPS) : 1.0f;
  out[i] = fminf(v, hi);
}

// ---------------------------------------------------------------------------
// K7: scatter 1.0 at observed (masked) indices when training.
// ---------------------------------------------------------------------------
__global__ void __launch_bounds__(32) set_observed(float* __restrict__ out,
                                                   const int* __restrict__ obs_idx,
                                                   const unsigned char* __restrict__ obs_mask,
                                                   const int* __restrict__ train_mask,
                                                   int n_ent) {
  if (train_mask[0] == 0) return;
  int h = blockIdx.x, j = threadIdx.x;
  if (obs_mask[h * MAX_OBS + j])
    out[(size_t)h * n_ent + obs_idx[h * MAX_OBS + j]] = 1.0f;
}

// ---------------------------------------------------------------------------
extern "C" void kernel_launch(void* const* d_in, const int* in_sizes, int n_in,
                              void* d_out, int out_size, void* d_ws, size_t ws_size,
                              hipStream_t stream) {
  (void)n_in; (void)ws_size;
  const float* ent           = (const float*)d_in[0];
  const float* rel           = (const float*)d_in[1];
  const float* hv            = (const float*)d_in[2];
  const int* obs_idx         = (const int*)d_in[3];
  const unsigned char* omask = (const unsigned char*)d_in[4];  // jnp.bool_ -> 1 byte
  const int* rel_id          = (const int*)d_in[5];
  const int* train_mask      = (const int*)d_in[7];
  float* out = (float*)d_out;

  int n_ent = in_sizes[2];          // 100000
  int nh    = out_size / n_ent;     // 256 heads (derived: out is nh x n_ent)

  // Workspace layout (all tiny: ~260 KB)
  char* ws   = (char*)d_ws;
  int*   heads = (int*)ws;                    // nh ints
  float* mx    = (float*)(ws + 1024);
  float* sm    = (float*)(ws + 2048);
  float* scal  = (float*)(ws + 3072);
  __bf16* Q    = (__bf16*)(ws + 4096);        // nh x 512 bf16

  find_heads<<<1, 1024, 0, stream>>>(hv, heads, n_ent, nh);
  build_q<<<nh, RANK, 0, stream>>>(ent, rel, heads, rel_id, Q);
  gemm_score<<<n_ent / 16, 256, 0, stream>>>(ent, Q, out, n_ent, nh / 16);
  softmax_stats<<<nh, 256, 0, stream>>>(out, mx, sm, n_ent);
  compute_scaling<<<nh, 32, 0, stream>>>(out, obs_idx, omask, mx, sm, scal, n_ent);
  dim3 g((n_ent + 255) / 256, nh);
  finalize<<<g, 256, 0, stream>>>(out, mx, sm, scal, train_mask, n_ent);
  set_observed<<<nh, 32, 0, stream>>>(out, obs_idx, omask, train_mask, n_ent);
}
